// Block_38835094290730
// MI455X (gfx1250) — matrix operand order repaired and checked by reading, not compiled, not run on hardware
//
#include <hip/hip_runtime.h>
#include <hip/hip_bf16.h>
#include <cstdint>
#include <cstddef>

// ---------------------------------------------------------------------------
// Problem constants (match reference)
// ---------------------------------------------------------------------------
#define BDIM 8
#define CDIM 256
#define LDIM 512
#define NPER 6
#define KMAXT 17
#define HIDD 1024
#define EPSLN 1e-6f

typedef __bf16 bf16;
typedef __attribute__((ext_vector_type(16))) __bf16 v16bf;
typedef __attribute__((ext_vector_type(8)))  __bf16 v8bf;
typedef __attribute__((ext_vector_type(8)))  float  v8f;

__device__ __forceinline__ float gelu_exact(float x) {
    return 0.5f * x * (1.0f + erff(x * 0.70710678118654752440f));
}

// ---------------------------------------------------------------------------
// WMMA fragment loaders, following CDNA5 ISA 7.12.2 VGPR layouts (wave32).
//
// A (16x32 bf16, M x K), A stored row-major [M,K] (lda = K):
//   lane<16 : row M=lane,    halves 0..7 -> K=kb+0..7,  halves 8..15 -> K=kb+16..23
//   lane>=16: row M=lane-16, halves 0..7 -> K=kb+8..15, halves 8..15 -> K=kb+24..31
// -> two contiguous 16B loads per fragment.
//
// B (32x16 bf16, K x N), activation stored row-major [N,K] (ldb = K):
//   lane<16 : col N=lane,    halves 0..7 -> K=kb+0..7,  halves 8..15 -> K=kb+16..23
//   lane>=16: col N=lane-16, halves 0..7 -> K=kb+8..15, halves 8..15 -> K=kb+24..31
// -> two contiguous 16B loads per fragment.
// ---------------------------------------------------------------------------
__device__ __forceinline__ v16bf load_a_frag(const bf16* __restrict__ A, int lda,
                                             int mo, int kb, int lane) {
    const int m  = lane & 15;
    const int ks = (lane < 16) ? 0 : 8;
    const bf16* p = A + (size_t)(mo + m) * lda + kb + ks;
    v8bf lo = *(const v8bf*)(p);
    v8bf hi = *(const v8bf*)(p + 16);
    v16bf r;
#pragma unroll
    for (int i = 0; i < 8; ++i) { r[i] = lo[i]; r[i + 8] = hi[i]; }
    return r;
}

__device__ __forceinline__ v16bf load_b_frag(const bf16* __restrict__ Bm, int ldb,
                                             int no, int kb, int lane) {
    const int n  = lane & 15;
    const int ks = (lane < 16) ? 0 : 8;
    const bf16* p = Bm + (size_t)(no + n) * ldb + kb + ks;
    v8bf lo = *(const v8bf*)(p);
    v8bf hi = *(const v8bf*)(p + 16);
    v16bf r;
#pragma unroll
    for (int i = 0; i < 8; ++i) { r[i] = lo[i]; r[i + 8] = hi[i]; }
    return r;
}

// ---------------------------------------------------------------------------
// Generic batched WMMA GEMM:  D[b] = A[M,K] x B[b][K,N] (+bias, epilogue)
//   A        : bf16 [M,K] row-major, shared across batch
//   Bm       : bf16 [batch][N,K] row-major ("transposed" activation layout)
//   outF     : f32  [batch][M,N]
//   outT     : bf16 [batch][N,M]
// Modes: 0: outF = d+bias
//        1: outT = bf16(gelu(d+bias))
//        3: outF += d+bias                        (residual accumulate)
//        4: outF = tF[b,n,m] + g2[m]*(d+bias)     (final MLP residual -> d_out)
//
// Block = 128 threads = 4 waves stacked in M. Each wave owns a 32x64 tile
// (acc[2][4]): per K-step of 32 it issues 12x global_load_b128 and 8x
// v_wmma_f32_16x16x32_bf16 (1.5 loads/WMMA).
// Requires M%128==0, N%64==0, K%32==0 (true for all GEMMs here:
// M in {256, 256*Ktap, 1024}, N=512, K in {256,1024}).
// ---------------------------------------------------------------------------
__global__ __launch_bounds__(128)
void gemm_wmma(const bf16* __restrict__ A, const bf16* __restrict__ Bm,
               const float* __restrict__ bias,
               float* __restrict__ outF, bf16* __restrict__ outT,
               const float* __restrict__ tF, const float* __restrict__ g2,
               int M, int N, int K, int mode)
{
    const int lane = threadIdx.x & 31;
    const int wave = threadIdx.x >> 5;
    const int mo = blockIdx.x * 128 + wave * 32;   // wave tile: 32 rows
    const int no = blockIdx.y * 64;                // wave tile: 64 cols
    const int bat = blockIdx.z;

    const bf16* __restrict__ Bb = Bm + (size_t)bat * N * K;

    v8f acc[2][4];
#pragma unroll
    for (int mi = 0; mi < 2; ++mi)
#pragma unroll
        for (int ni = 0; ni < 4; ++ni)
#pragma unroll
            for (int r = 0; r < 8; ++r) acc[mi][ni][r] = 0.0f;

    for (int kb = 0; kb < K; kb += 32) {
        v16bf a0 = load_a_frag(A, K, mo,      kb, lane);
        v16bf a1 = load_a_frag(A, K, mo + 16, kb, lane);
        v16bf bfr[4];
#pragma unroll
        for (int ni = 0; ni < 4; ++ni)
            bfr[ni] = load_b_frag(Bb, K, no + ni * 16, kb, lane);
#pragma unroll
        for (int ni = 0; ni < 4; ++ni) {
            acc[0][ni] = __builtin_amdgcn_wmma_f32_16x16x32_bf16(false, a0, false, bfr[ni],
                                                                 (short)0, acc[0][ni], false, false);
            acc[1][ni] = __builtin_amdgcn_wmma_f32_16x16x32_bf16(false, a1, false, bfr[ni],
                                                                 (short)0, acc[1][ni], false, false);
        }
    }

    // C/D layout: VGPR r, lanes 0-15 -> M = r, N = lane; lanes 16-31 -> M = 8+r.
    const int nl = lane & 15;
    const int mh = (lane < 16) ? 0 : 8;
#pragma unroll
    for (int mi = 0; mi < 2; ++mi) {
#pragma unroll
        for (int ni = 0; ni < 4; ++ni) {
#pragma unroll
            for (int r = 0; r < 8; ++r) {
                const int m = mo + mi * 16 + mh + r;
                const int n = no + ni * 16 + nl;
                float d = acc[mi][ni][r] + bias[m];
                if (mode == 0) {
                    outF[((size_t)bat * M + m) * N + n] = d;
                } else if (mode == 1) {
                    outT[((size_t)bat * N + n) * M + m] = (bf16)gelu_exact(d);
                } else if (mode == 3) {
                    outF[((size_t)bat * M + m) * N + n] += d;
                } else { // mode 4: final residual, write [B, M=C, N=L]
                    const size_t ti = ((size_t)bat * N + n) * M + m;
                    outF[((size_t)bat * M + m) * N + n] = tF[ti] + g2[m] * d;
                }
            }
        }
    }
}

// ---------------------------------------------------------------------------
// Elementwise / data-movement kernels
// ---------------------------------------------------------------------------
__global__ __launch_bounds__(256)
void f32_to_bf16_kernel(const float* __restrict__ src, bf16* __restrict__ dst, size_t n) {
    size_t i = (size_t)blockIdx.x * 256 + threadIdx.x;
    if (i < n) dst[i] = (bf16)src[i];
}

// x [B,C,L] f32 -> xT [B,L,C] bf16
__global__ __launch_bounds__(256)
void transpose_x_kernel(const float* __restrict__ x, bf16* __restrict__ xT) {
    size_t idx = (size_t)blockIdx.x * 256 + threadIdx.x;
    if (idx >= (size_t)BDIM * CDIM * LDIM) return;
    const int l = (int)(idx % LDIM);
    const int c = (int)((idx / LDIM) % CDIM);
    const int b = (int)(idx / ((size_t)CDIM * LDIM));
    xT[((size_t)b * LDIM + l) * CDIM + c] = (bf16)x[idx];
}

// prodT[b,l,c] = bf16(a[b,c,l] * v[b,c,l])
__global__ __launch_bounds__(256)
void prod_kernel(const float* __restrict__ a, const float* __restrict__ v,
                 bf16* __restrict__ prodT) {
    size_t idx = (size_t)blockIdx.x * 256 + threadIdx.x;
    if (idx >= (size_t)BDIM * CDIM * LDIM) return;
    const int l = (int)(idx % LDIM);
    const int c = (int)((idx / LDIM) % CDIM);
    const int b = (int)(idx / ((size_t)CDIM * LDIM));
    prodT[((size_t)b * LDIM + l) * CDIM + c] = (bf16)(a[idx] * v[idx]);
}

// ---------------------------------------------------------------------------
// DCNv3-1D core: softmax over K taps + linear-interp gather, per (b,c,l).
// off/ml are [B, C*K, L]; v is [B,C,L]; result stored transposed bf16 [B,L,C].
// Templated on K so tap arrays stay in registers.
// ---------------------------------------------------------------------------
template <int K>
__global__ __launch_bounds__(256)
void dcn_kernel(const float* __restrict__ v, const float* __restrict__ off,
                const float* __restrict__ ml, bf16* __restrict__ sT) {
    size_t idx = (size_t)blockIdx.x * 256 + threadIdx.x;
    if (idx >= (size_t)BDIM * CDIM * LDIM) return;
    const int l = (int)(idx % LDIM);
    const int c = (int)((idx / LDIM) % CDIM);
    const int b = (int)(idx / ((size_t)CDIM * LDIM));

    const float* __restrict__ vrow = v + ((size_t)b * CDIM + c) * LDIM;
    const size_t rowbase = (((size_t)b * CDIM + c) * K) * LDIM + l;

    float mv[K], ov[K];
    float mmax = -3.0e38f;
#pragma unroll
    for (int k = 0; k < K; ++k) {
        mv[k] = ml[rowbase + (size_t)k * LDIM];
        ov[k] = off[rowbase + (size_t)k * LDIM];
        mmax = fmaxf(mmax, mv[k]);
    }
    float ssum = 0.0f;
#pragma unroll
    for (int k = 0; k < K; ++k) { mv[k] = expf(mv[k] - mmax); ssum += mv[k]; }
    const float inv = 1.0f / ssum;

    const float center = 0.5f * (float)(K - 1);
    float acc = 0.0f;
#pragma unroll
    for (int k = 0; k < K; ++k) {
        const float p  = (float)l + ((float)k - center) + ov[k];
        const float p0 = floorf(p);
        const float w  = p - p0;
        const int   i0 = (int)p0;
        const int   i1 = i0 + 1;
        const float g0 = (i0 >= 0 && i0 < LDIM) ? vrow[i0] : 0.0f;
        const float g1 = (i1 >= 0 && i1 < LDIM) ? vrow[i1] : 0.0f;
        acc += (mv[k] * inv) * ((1.0f - w) * g0 + w * g1);
    }
    sT[((size_t)b * LDIM + l) * CDIM + c] = (bf16)acc;
}

// ---------------------------------------------------------------------------
// Residual + LayerNorm over C=256 channels at each (b,l).
//   t[b,l,c]  = x[b,c,l] + ls[c]*res[b,c,l]      -> tF  (f32, [B,L,C])
//   tn[b,l,c] = LN(t)*g + beta                   -> tn  (bf16, [B,L,C])
// One 256-thread block per (b,l).
// ---------------------------------------------------------------------------
__global__ __launch_bounds__(256)
void ln_kernel(const float* __restrict__ x, const float* __restrict__ res,
               const float* __restrict__ ls, const float* __restrict__ g,
               const float* __restrict__ beta,
               float* __restrict__ tF, bf16* __restrict__ tn) {
    const int bl = blockIdx.x;
    const int b = bl / LDIM;
    const int l = bl % LDIM;
    const int c = threadIdx.x;

    const size_t xi = ((size_t)b * CDIM + c) * LDIM + l;
    const float val = x[xi] + ls[c] * res[xi];

    __shared__ float red[256];
    red[c] = val;
    __syncthreads();
#pragma unroll
    for (int s = 128; s > 0; s >>= 1) {
        if (c < s) red[c] += red[c + s];
        __syncthreads();
    }
    const float mu = red[0] * (1.0f / CDIM);
    __syncthreads();
    const float d = val - mu;
    red[c] = d * d;
    __syncthreads();
#pragma unroll
    for (int s = 128; s > 0; s >>= 1) {
        if (c < s) red[c] += red[c + s];
        __syncthreads();
    }
    const float var = red[0] * (1.0f / CDIM);

    const float tnv = d * rsqrtf(var + EPSLN) * g[c] + beta[c];
    const size_t o = ((size_t)b * LDIM + l) * CDIM + c;
    tF[o] = val;
    tn[o] = (bf16)tnv;
}

// ---------------------------------------------------------------------------
// Host-side launcher
// ---------------------------------------------------------------------------
extern "C" void kernel_launch(void* const* d_in, const int* in_sizes, int n_in,
                              void* d_out, int out_size, void* d_ws, size_t ws_size,
                              hipStream_t stream) {
    (void)in_sizes; (void)n_in; (void)out_size; (void)ws_size;

    const float* x    = (const float*)d_in[0];
    const float* Wa   = (const float*)d_in[1];
    const float* ba   = (const float*)d_in[2];
    const float* Wvd  = (const float*)d_in[3];
    const float* bvd  = (const float*)d_in[4];
    const float* Woff = (const float*)d_in[5];
    const float* boff = (const float*)d_in[6];
    const float* Wm   = (const float*)d_in[7];
    const float* bm   = (const float*)d_in[8];
    const float* Wod  = (const float*)d_in[9];
    const float* bod  = (const float*)d_in[10];
    const float* Wv   = (const float*)d_in[11];
    const float* bv   = (const float*)d_in[12];
    const float* Wp   = (const float*)d_in[13];
    const float* bp   = (const float*)d_in[14];
    const float* ls   = (const float*)d_in[15];
    const float* g2   = (const float*)d_in[16];
    const float* ln_g = (const float*)d_in[17];
    const float* ln_b = (const float*)d_in[18];
    const float* W1   = (const float*)d_in[19];
    const float* b1   = (const float*)d_in[20];
    const float* W2   = (const float*)d_in[21];
    const float* b2   = (const float*)d_in[22];

    float* out = (float*)d_out;

    // ---- bump allocator over d_ws ----
    char*  base = (char*)d_ws;
    size_t cur  = 0;
    auto alloc = [&](size_t bytes) -> void* {
        void* p = base + cur;
        cur += (bytes + 255) & ~(size_t)255;
        return p;
    };

    const size_t nBCL = (size_t)BDIM * CDIM * LDIM;          // 1,048,576
    const size_t nCC  = (size_t)NPER * CDIM * CDIM;          // all-branch 256x256 weights
    const size_t nCKC = (size_t)NPER * CDIM * KMAXT * CDIM;  // all-branch offset/mask weights

    // bf16 weights
    bf16* Wa_bf  = (bf16*)alloc(nCC  * 2);
    bf16* Wvd_bf = (bf16*)alloc(nCC  * 2);
    bf16* Woff_bf= (bf16*)alloc(nCKC * 2);
    bf16* Wm_bf  = (bf16*)alloc(nCKC * 2);
    bf16* Wod_bf = (bf16*)alloc(nCC  * 2);
    bf16* Wv_bf  = (bf16*)alloc(nCC  * 2);
    bf16* Wp_bf  = (bf16*)alloc(nCC  * 2);
    bf16* W1_bf  = (bf16*)alloc((size_t)HIDD * CDIM * 2);
    bf16* W2_bf  = (bf16*)alloc((size_t)CDIM * HIDD * 2);

    // activations
    bf16*  xT    = (bf16*)alloc(nBCL * 2);                   // [B,L,C]
    bf16*  gT    = (bf16*)alloc(nBCL * 2);                   // [B,L,C]
    bf16*  sT    = (bf16*)alloc(nBCL * 2);                   // [B,L,C]
    bf16*  prodT = (bf16*)alloc(nBCL * 2);                   // [B,L,C]
    bf16*  tn    = (bf16*)alloc(nBCL * 2);                   // [B,L,C]
    bf16*  hT    = (bf16*)alloc((size_t)BDIM * LDIM * HIDD * 2); // [B,L,HID]
    float* vbuf  = (float*)alloc(nBCL * 4);                  // [B,C,L]
    float* vvbuf = (float*)alloc(nBCL * 4);                  // [B,C,L]
    float* outd  = (float*)alloc(nBCL * 4);                  // [B,C,L]
    float* resb  = (float*)alloc(nBCL * 4);                  // [B,C,L]
    float* tF    = (float*)alloc(nBCL * 4);                  // [B,L,C]
    float* offb  = (float*)alloc((size_t)BDIM * CDIM * KMAXT * LDIM * 4); // [B,C*K,L]
    float* mlb   = (float*)alloc((size_t)BDIM * CDIM * KMAXT * LDIM * 4); // [B,C*K,L]

    // ---- weight conversions (f32 -> bf16) ----
    auto conv = [&](const float* s, bf16* d, size_t n) {
        f32_to_bf16_kernel<<<dim3((unsigned)((n + 255) / 256)), dim3(256), 0, stream>>>(s, d, n);
    };
    conv(Wa,   Wa_bf,   nCC);
    conv(Wvd,  Wvd_bf,  nCC);
    conv(Woff, Woff_bf, nCKC);
    conv(Wm,   Wm_bf,   nCKC);
    conv(Wod,  Wod_bf,  nCC);
    conv(Wv,   Wv_bf,   nCC);
    conv(Wp,   Wp_bf,   nCC);
    conv(W1,   W1_bf,   (size_t)HIDD * CDIM);
    conv(W2,   W2_bf,   (size_t)CDIM * HIDD);

    // x -> xT (bf16, [B,L,C]); zero the residual accumulator
    transpose_x_kernel<<<dim3((unsigned)((nBCL + 255) / 256)), dim3(256), 0, stream>>>(x, xT);
    hipMemsetAsync(resb, 0, nBCL * 4, stream);

    const dim3 blk128(128), blk256(256);
    const dim3 grdElem((unsigned)((nBCL + 255) / 256));
    auto gemm = [&](const bf16* A, const bf16* B, const float* bias, float* oF, bf16* oT,
                    const float* tf, const float* gg, int M, int N, int K, int mode) {
        gemm_wmma<<<dim3((unsigned)(M / 128), (unsigned)(N / 64), BDIM), blk128, 0, stream>>>(
            A, B, bias, oF, oT, tf, gg, M, N, K, mode);
    };

    // ---- the 6 deformable branches ----
    for (int i = 0; i < NPER; ++i) {
        const int Ktap = 7 + 2 * i;
        const int CK   = CDIM * Ktap;
        const bf16* WaI   = Wa_bf   + (size_t)i * CDIM * CDIM;
        const bf16* WvdI  = Wvd_bf  + (size_t)i * CDIM * CDIM;
        const bf16* WoffI = Woff_bf + (size_t)i * CDIM * KMAXT * CDIM;
        const bf16* WmI   = Wm_bf   + (size_t)i * CDIM * KMAXT * CDIM;
        const bf16* WodI  = Wod_bf  + (size_t)i * CDIM * CDIM;
        const bf16* WvI   = Wv_bf   + (size_t)i * CDIM * CDIM;
        const bf16* WpI   = Wp_bf   + (size_t)i * CDIM * CDIM;
        const float* baI   = ba   + (size_t)i * CDIM;
        const float* bvdI  = bvd  + (size_t)i * CDIM;
        const float* boffI = boff + (size_t)i * CDIM * KMAXT;
        const float* bmI   = bm   + (size_t)i * CDIM * KMAXT;
        const float* bodI  = bod  + (size_t)i * CDIM;
        const float* bvI   = bv   + (size_t)i * CDIM;
        const float* bpI   = bp   + (size_t)i * CDIM;

        // g = gelu(Wa x + ba), stored transposed bf16
        gemm(WaI, xT, baI, nullptr, gT, nullptr, nullptr, CDIM, LDIM, CDIM, 1);
        // v = Wvd g + bvd  [B,C,L]
        gemm(WvdI, gT, bvdI, vbuf, nullptr, nullptr, nullptr, CDIM, LDIM, CDIM, 0);
        // off / mask logits  [B, C*K, L]
        gemm(WoffI, gT, boffI, offb, nullptr, nullptr, nullptr, CK, LDIM, CDIM, 0);
        gemm(WmI,   gT, bmI,   mlb,  nullptr, nullptr, nullptr, CK, LDIM, CDIM, 0);
        // deformable sampling -> sT bf16 [B,L,C]
        switch (Ktap) {
            case 7:  dcn_kernel<7> <<<grdElem, blk256, 0, stream>>>(vbuf, offb, mlb, sT); break;
            case 9:  dcn_kernel<9> <<<grdElem, blk256, 0, stream>>>(vbuf, offb, mlb, sT); break;
            case 11: dcn_kernel<11><<<grdElem, blk256, 0, stream>>>(vbuf, offb, mlb, sT); break;
            case 13: dcn_kernel<13><<<grdElem, blk256, 0, stream>>>(vbuf, offb, mlb, sT); break;
            case 15: dcn_kernel<15><<<grdElem, blk256, 0, stream>>>(vbuf, offb, mlb, sT); break;
            default: dcn_kernel<17><<<grdElem, blk256, 0, stream>>>(vbuf, offb, mlb, sT); break;
        }
        // out_d = Wod s + bod  [B,C,L]
        gemm(WodI, sT, bodI, outd, nullptr, nullptr, nullptr, CDIM, LDIM, CDIM, 0);
        // vv = Wv x + bv  [B,C,L]
        gemm(WvI, xT, bvI, vvbuf, nullptr, nullptr, nullptr, CDIM, LDIM, CDIM, 0);
        // prodT = bf16(out_d * vv)  [B,L,C]
        prod_kernel<<<grdElem, blk256, 0, stream>>>(outd, vvbuf, prodT);
        // res += Wp prod + bp
        gemm(WpI, prodT, bpI, resb, nullptr, nullptr, nullptr, CDIM, LDIM, CDIM, 3);
    }

    // ---- residual + LayerNorm ----
    ln_kernel<<<dim3(BDIM * LDIM), blk256, 0, stream>>>(x, resb, ls, ln_g, ln_b, tF, tn);

    // ---- MLP: h = gelu(tn W1^T + b1); out = t + g2*(h W2^T + b2), back to [B,C,L]
    gemm(W1_bf, tn, b1, nullptr, hT, nullptr, nullptr, HIDD, LDIM, CDIM, 1);
    gemm(W2_bf, hT, b2, out, nullptr, tF, g2, CDIM, LDIM, HIDD, 4);
}
